// VisionMamba_60490319396874
// MI455X (gfx1250) — matrix operand order repaired and checked by reading, not compiled, not run on hardware
//
#include <hip/hip_runtime.h>
#include <hip/hip_bf16.h>

#define DEV __device__ __forceinline__

typedef __attribute__((ext_vector_type(16))) __bf16       v16bf;
typedef __attribute__((ext_vector_type(8)))  float        v8f;
typedef __attribute__((ext_vector_type(4)))  unsigned int v4u;

#define BATCH   32
#define LSEQ    197
#define DMODEL  384
#define EINNER  768
#define NLAYERS 12
#define ROWS    (BATCH * LSEQ)     /* 6304 */
#define NPATCH  196
#define PROWS   (BATCH * NPATCH)   /* 6272 */
#define KPATCH  768
#define DTRANK  24
#define DTPAD   32
#define NSTATE  16
#define DBLW    56
#define DCONV   4

DEV unsigned short f32_to_bf16(float f) {
  unsigned int u = __builtin_bit_cast(unsigned int, f);
  u += 0x7fffu + ((u >> 16) & 1u);
  return (unsigned short)(u >> 16);
}
DEV float silu_f(float x)     { return x / (1.0f + expf(-x)); }
DEV float softplus_f(float x) { return x > 20.0f ? x : log1pf(expf(x)); }
DEV v4u  zero4() { v4u z; z[0] = 0u; z[1] = 0u; z[2] = 0u; z[3] = 0u; return z; }

// --- gfx1250 async global->LDS copy (ASYNCcnt-tracked, no VGPR round trip) ---
DEV void async_to_lds_b128(unsigned lds_addr, const unsigned short* gptr) {
  unsigned long long ga = (unsigned long long)(size_t)gptr;
  asm volatile("global_load_async_to_lds_b128 %0, %1, off"
               :: "v"(lds_addr), "v"(ga) : "memory");
}
DEV void wait_asynccnt0() { asm volatile("s_wait_asynccnt 0x0" ::: "memory"); }
DEV void wait_asynccnt1() { asm volatile("s_wait_asynccnt 0x1" ::: "memory"); }

// ---------------------------------------------------------------------------
// fp32 -> bf16 conversion kernels
// ---------------------------------------------------------------------------
__global__ void cvt_f32_bf16(const float* __restrict__ in,
                             unsigned short* __restrict__ out, long long n) {
  long long i = (long long)blockIdx.x * blockDim.x + threadIdx.x;
  if (i < n) out[i] = f32_to_bf16(in[i]);
}

// rows x 32 output, first `valid` columns from in (row stride in_stride), rest 0
__global__ void pad_bf16(const float* __restrict__ in,
                         unsigned short* __restrict__ out,
                         int rows, int in_stride, int valid) {
  long long i = (long long)blockIdx.x * blockDim.x + threadIdx.x;
  if (i >= (long long)rows * DTPAD) return;
  int k = (int)(i & (DTPAD - 1));
  long long r = i >> 5;
  out[i] = (k < valid) ? f32_to_bf16(in[r * in_stride + k]) : (unsigned short)0;
}

// ---------------------------------------------------------------------------
// im2col for 16x16/stride-16 patch embed, direct to bf16. K order = c*256+py*16+px
// ---------------------------------------------------------------------------
__global__ void im2col_bf16(const float* __restrict__ x,
                            unsigned short* __restrict__ out) {
  long long i = (long long)blockIdx.x * blockDim.x + threadIdx.x;
  if (i >= (long long)PROWS * KPATCH) return;
  int k   = (int)(i % KPATCH);
  int row = (int)(i / KPATCH);
  int b  = row / NPATCH, pidx = row % NPATCH;
  int ph = pidx / 14,    pw   = pidx % 14;
  int c  = k / 256,      rem  = k % 256;
  int py = rem / 16,     px   = rem % 16;
  float v = x[(((long long)b * 3 + c) * 224 + (ph * 16 + py)) * 224 + (pw * 16 + px)];
  out[i] = f32_to_bf16(v);
}

// hs[b,l,:] = (l<196 ? patchGEMM(b,l) : cls_token), res = 0
__global__ void assemble_tokens(const float* __restrict__ ptmp,
                                const float* __restrict__ cls_tok,
                                float* __restrict__ hs, float* __restrict__ res) {
  long long i = (long long)blockIdx.x * blockDim.x + threadIdx.x;
  if (i >= (long long)ROWS * DMODEL) return;
  int d   = (int)(i % DMODEL);
  int row = (int)(i / DMODEL);
  int b = row / LSEQ, l = row % LSEQ;
  float v = (l < NPATCH) ? ptmp[((long long)b * NPATCH + l) * DMODEL + d] : cls_tok[d];
  hs[i]  = v;
  res[i] = 0.0f;
}

// ---------------------------------------------------------------------------
// res += hs, then LayerNorm(res) -> bf16 (GEMM A operand). One block per row.
// ---------------------------------------------------------------------------
__global__ __launch_bounds__(128) void add_ln_bf16(
    float* __restrict__ res, const float* __restrict__ hs,
    const float* __restrict__ g, const float* __restrict__ bta,
    unsigned short* __restrict__ outbf) {
  __shared__ float s0[128], s1[128];
  int row = blockIdx.x, tid = threadIdx.x;
  float* rp = res + (long long)row * DMODEL;
  const float* hp = hs + (long long)row * DMODEL;
  float vals[3], sum = 0.f, sq = 0.f;
#pragma unroll
  for (int i = 0; i < 3; ++i) {
    int d = tid + i * 128;
    float v = rp[d] + hp[d];
    rp[d] = v; vals[i] = v;
    sum += v; sq += v * v;
  }
  s0[tid] = sum; s1[tid] = sq; __syncthreads();
  for (int off = 64; off > 0; off >>= 1) {
    if (tid < off) { s0[tid] += s0[tid + off]; s1[tid] += s1[tid + off]; }
    __syncthreads();
  }
  float mean = s0[0] * (1.0f / DMODEL);
  float var  = s1[0] * (1.0f / DMODEL) - mean * mean;
  float rstd = rsqrtf(var + 1e-5f);
#pragma unroll
  for (int i = 0; i < 3; ++i) {
    int d = tid + i * 128;
    float v = (vals[i] - mean) * rstd * g[d] + bta[d];
    outbf[(long long)row * DMODEL + d] = f32_to_bf16(v);
  }
}

// Final LN on the cls row (l = 196) of res+hs -> bf16 [BATCH, DMODEL]
__global__ __launch_bounds__(128) void final_ln_cls(
    const float* __restrict__ res, const float* __restrict__ hs,
    const float* __restrict__ g, const float* __restrict__ bta,
    unsigned short* __restrict__ outbf) {
  __shared__ float s0[128], s1[128];
  int b = blockIdx.x, tid = threadIdx.x;
  long long row = (long long)b * LSEQ + (LSEQ - 1);
  const float* rp = res + row * DMODEL;
  const float* hp = hs + row * DMODEL;
  float vals[3], sum = 0.f, sq = 0.f;
#pragma unroll
  for (int i = 0; i < 3; ++i) {
    int d = tid + i * 128;
    float v = rp[d] + hp[d];
    vals[i] = v; sum += v; sq += v * v;
  }
  s0[tid] = sum; s1[tid] = sq; __syncthreads();
  for (int off = 64; off > 0; off >>= 1) {
    if (tid < off) { s0[tid] += s0[tid + off]; s1[tid] += s1[tid + off]; }
    __syncthreads();
  }
  float mean = s0[0] * (1.0f / DMODEL);
  float var  = s1[0] * (1.0f / DMODEL) - mean * mean;
  float rstd = rsqrtf(var + 1e-5f);
#pragma unroll
  for (int i = 0; i < 3; ++i) {
    int d = tid + i * 128;
    float v = (vals[i] - mean) * rstd * g[d] + bta[d];
    outbf[(long long)b * DMODEL + d] = f32_to_bf16(v);
  }
}

// ---------------------------------------------------------------------------
// Depthwise causal conv (width 4) + SiLU over xi = xz[:, :768].
// ---------------------------------------------------------------------------
__global__ void conv_silu(const float* __restrict__ xz,
                          const float* __restrict__ cw,
                          const float* __restrict__ cb,
                          float* __restrict__ xc,
                          unsigned short* __restrict__ xcbf) {
  long long i = (long long)blockIdx.x * blockDim.x + threadIdx.x;
  if (i >= (long long)ROWS * EINNER) return;
  int e   = (int)(i % EINNER);
  int row = (int)(i / EINNER);
  int b = row / LSEQ, l = row % LSEQ;
  float acc = cb[e];
#pragma unroll
  for (int j = 0; j < DCONV; ++j) {
    int li = l - (DCONV - 1) + j;
    if (li >= 0)
      acc += xz[((long long)b * LSEQ + li) * (2 * EINNER) + e] * cw[e * DCONV + j];
  }
  float v = silu_f(acc);
  xc[i]   = v;
  xcbf[i] = f32_to_bf16(v);
}

// ---------------------------------------------------------------------------
// Selective scan: one thread per (b, e) channel, 16-state in registers.
// ---------------------------------------------------------------------------
__global__ __launch_bounds__(256) void selective_scan(
    const float* __restrict__ dtv, const float* __restrict__ xc,
    const float* __restrict__ dbl, const float* __restrict__ xz,
    const float* __restrict__ A_log, const float* __restrict__ D_skip,
    unsigned short* __restrict__ ybf) {
  int idx = blockIdx.x * blockDim.x + threadIdx.x;
  int e = idx % EINNER, b = idx / EINNER;
  if (b >= BATCH) return;
  float Arow[NSTATE];
#pragma unroll
  for (int n = 0; n < NSTATE; ++n) Arow[n] = -expf(A_log[(long long)e * NSTATE + n]);
  float dsk = D_skip[e];
  float h[NSTATE];
#pragma unroll
  for (int n = 0; n < NSTATE; ++n) h[n] = 0.f;
  for (int l = 0; l < LSEQ; ++l) {
    long long row = (long long)b * LSEQ + l;
    float dt = dtv[row * EINNER + e];
    float u  = xc [row * EINNER + e];
    const float* bc = dbl + row * DBLW;
    float y = 0.f;
#pragma unroll
    for (int n = 0; n < NSTATE; ++n) {
      float dA  = expf(dt * Arow[n]);
      float dBu = dt * bc[DTRANK + n] * u;
      h[n] = dA * h[n] + dBu;
      y += h[n] * bc[DTRANK + NSTATE + n];
    }
    y += dsk * u;
    float z = xz[row * (2 * EINNER) + EINNER + e];
    y *= silu_f(z);
    ybf[row * EINNER + e] = f32_to_bf16(y);
  }
}

// ---------------------------------------------------------------------------
// bf16 WMMA GEMM: C[M,N] = A[M,K](bf16,row) x W[N,K](bf16,row)^T, f32 accum.
// Block = 8 waves, tile 128(M) x 64(N). Interior blocks: async global->LDS
// double-buffered W staging (ASYNCcnt), no predication. Edge blocks: checked.
// ---------------------------------------------------------------------------
DEV void wmma_step(const unsigned short* tile, const v16bf& afrag,
                   v8f acc[4], int l15, int half) {
#pragma unroll
  for (int s = 0; s < 4; ++s) {
    v16bf bfrag;
    const unsigned short* bp = tile + (s * 16 + l15) * 32 + half * 8;
    ((v4u*)&bfrag)[0] = *(const v4u*)(bp);
    ((v4u*)&bfrag)[1] = *(const v4u*)(bp + 16);
    acc[s] = __builtin_amdgcn_wmma_f32_16x16x32_bf16(
        false, afrag, false, bfrag, (short)0, acc[s], false, false);
  }
}

__global__ __launch_bounds__(256) void gemm_bf16_wmma(
    const unsigned short* __restrict__ A, const unsigned short* __restrict__ W,
    float* __restrict__ C, int M, int N, int K,
    const float* __restrict__ bias, int act) {
  __shared__ unsigned short ldsB[2][64 * 32];  // double-buffered 64n x 32k tiles
  const int tid = threadIdx.x;
  const int wave = tid >> 5, lane = tid & 31;
  const int half = lane >> 4, l15 = lane & 15;
  const int blockN = blockIdx.x * 64;
  const int blockM = blockIdx.y * 128;
  const int mwave  = blockM + wave * 16;
  const int ldn = tid >> 2;      // 0..63  cooperative W-tile row
  const int ldc4 = tid & 3;      // 0..3   16-byte chunk within row

  v8f acc[4];
#pragma unroll
  for (int s = 0; s < 4; ++s)
#pragma unroll
    for (int r = 0; r < 8; ++r) acc[s][r] = 0.f;

  const int ntiles = K >> 5;
  const bool full = (blockN + 64 <= N) && (blockM + 128 <= M);

  if (full) {
    // ---------------- fast path: async double-buffered, unpredicated -------
    const unsigned short* wp = W + (long long)(blockN + ldn) * K + ldc4 * 8;
    const unsigned short* ap = A + (long long)(mwave + l15) * K;
    const unsigned lds_w0 = (unsigned)(size_t)&ldsB[0][ldn * 32 + ldc4 * 8];
    const unsigned lds_w1 = (unsigned)(size_t)&ldsB[1][ldn * 32 + ldc4 * 8];

    async_to_lds_b128(lds_w0, wp);  // prefetch tile 0
    for (int t = 0; t < ntiles; ++t) {
      const int cur = t & 1;
      // A fragment (synchronous, overlaps with async wait)
      v16bf afrag;
      {
        const unsigned short* a0 = ap + t * 32;
        ((v4u*)&afrag)[0] = *(const v4u*)(a0 + half * 8);
        ((v4u*)&afrag)[1] = *(const v4u*)(a0 + 16 + half * 8);
      }
      if (t + 1 < ntiles) {
        async_to_lds_b128(cur ? lds_w0 : lds_w1, wp + (t + 1) * 32);
        wait_asynccnt1();           // tile t done, prefetch may remain in flight
      } else {
        wait_asynccnt0();
      }
      __syncthreads();              // tile t visible to all waves
      wmma_step(&ldsB[cur][0], afrag, acc, l15, half);
      __syncthreads();              // all waves done with buf[cur]
    }
  } else {
    // ---------------- edge path: bounds-checked, single buffer -------------
    for (int k0 = 0; k0 < K; k0 += 32) {
      {
        int gn = blockN + ldn;
        v4u d = zero4();
        if (gn < N) d = *(const v4u*)(W + (long long)gn * K + k0 + ldc4 * 8);
        *(v4u*)(&ldsB[0][ldn * 32 + ldc4 * 8]) = d;
      }
      __syncthreads();
      v16bf afrag;
      {
        int m = mwave + l15;
        v4u c0 = zero4(), c1 = zero4();
        if (m < M) {
          const unsigned short* ap = A + (long long)m * K + k0;
          c0 = *(const v4u*)(ap + half * 8);
          c1 = *(const v4u*)(ap + 16 + half * 8);
        }
        ((v4u*)&afrag)[0] = c0;
        ((v4u*)&afrag)[1] = c1;
      }
      wmma_step(&ldsB[0][0], afrag, acc, l15, half);
      __syncthreads();
    }
  }

  // epilogue: VGPR r -> M = r (lanes 0-15) / 8+r (lanes 16-31), N = l15
#pragma unroll
  for (int s = 0; s < 4; ++s) {
    int n = blockN + s * 16 + l15;
    if (n >= N) continue;
    float bv = bias ? bias[n] : 0.f;
#pragma unroll
    for (int r = 0; r < 8; ++r) {
      int m = mwave + half * 8 + r;
      if (m < M) {
        float v = acc[s][r] + bv;
        if (act == 1) v = softplus_f(v);
        C[(long long)m * N + n] = v;
      }
    }
  }
}

// ---------------------------------------------------------------------------
// host launcher
// ---------------------------------------------------------------------------
extern "C" void kernel_launch(void* const* d_in, const int* in_sizes, int n_in,
                              void* d_out, int out_size, void* d_ws, size_t ws_size,
                              hipStream_t stream) {
  (void)in_sizes; (void)n_in; (void)out_size; (void)ws_size;
  const float* x       = (const float*)d_in[0];
  const float* patch_w = (const float*)d_in[1];
  const float* patch_b = (const float*)d_in[2];
  const float* cls_tok = (const float*)d_in[3];
  const float* ln_g    = (const float*)d_in[4];
  const float* ln_b    = (const float*)d_in[5];
  const float* in_w    = (const float*)d_in[6];
  const float* conv_w  = (const float*)d_in[7];
  const float* conv_b  = (const float*)d_in[8];
  const float* xproj_w = (const float*)d_in[9];
  const float* dt_w    = (const float*)d_in[10];
  const float* dt_b    = (const float*)d_in[11];
  const float* A_log   = (const float*)d_in[12];
  const float* D_skip  = (const float*)d_in[13];
  const float* out_w   = (const float*)d_in[14];
  const float* bb_g    = (const float*)d_in[15];
  const float* bb_b    = (const float*)d_in[16];
  const float* head_w  = (const float*)d_in[17];
  float* out = (float*)d_out;

  char* p = (char*)d_ws;
  auto carve = [&](size_t bytes) -> char* {
    char* q = p; p += (bytes + 255) & ~(size_t)255; return q;
  };
  float* res  = (float*)carve((size_t)ROWS * DMODEL * 4);
  float* hs   = (float*)carve((size_t)ROWS * DMODEL * 4);
  float* xz   = (float*)carve((size_t)ROWS * 2 * EINNER * 4);
  float* xc   = (float*)carve((size_t)ROWS * EINNER * 4);
  float* dbl  = (float*)carve((size_t)ROWS * DBLW * 4);
  float* dtv  = (float*)carve((size_t)ROWS * EINNER * 4);
  float* ptmp = (float*)carve((size_t)PROWS * DMODEL * 4);
  unsigned short* abf    = (unsigned short*)carve((size_t)ROWS * EINNER * 2);
  unsigned short* wpatch = (unsigned short*)carve((size_t)DMODEL * KPATCH * 2);
  unsigned short* win    = (unsigned short*)carve((size_t)NLAYERS * 2 * EINNER * DMODEL * 2);
  unsigned short* wx     = (unsigned short*)carve((size_t)NLAYERS * DBLW * EINNER * 2);
  unsigned short* wdt    = (unsigned short*)carve((size_t)NLAYERS * EINNER * DTPAD * 2);
  unsigned short* wout   = (unsigned short*)carve((size_t)NLAYERS * DMODEL * EINNER * 2);
  unsigned short* whead  = (unsigned short*)carve((size_t)1000 * DMODEL * 2);

  auto cdiv = [](long long a, long long b) { return (int)((a + b - 1) / b); };

  // ---- weight conversion (bf16), one launch per tensor family ----
  { long long n = (long long)DMODEL * KPATCH;
    cvt_f32_bf16<<<cdiv(n, 256), 256, 0, stream>>>(patch_w, wpatch, n); }
  { long long n = (long long)NLAYERS * 2 * EINNER * DMODEL;
    cvt_f32_bf16<<<cdiv(n, 256), 256, 0, stream>>>(in_w, win, n); }
  { long long n = (long long)NLAYERS * DBLW * EINNER;
    cvt_f32_bf16<<<cdiv(n, 256), 256, 0, stream>>>(xproj_w, wx, n); }
  { int rows = NLAYERS * EINNER;  // dt_proj_w [*,24] zero-padded to K=32
    pad_bf16<<<cdiv((long long)rows * DTPAD, 256), 256, 0, stream>>>(dt_w, wdt, rows, DTRANK, DTRANK); }
  { long long n = (long long)NLAYERS * DMODEL * EINNER;
    cvt_f32_bf16<<<cdiv(n, 256), 256, 0, stream>>>(out_w, wout, n); }
  { long long n = (long long)1000 * DMODEL;
    cvt_f32_bf16<<<cdiv(n, 256), 256, 0, stream>>>(head_w, whead, n); }

  // ---- patch embed (im2col GEMM) + token assembly ----
  im2col_bf16<<<cdiv((long long)PROWS * KPATCH, 256), 256, 0, stream>>>(x, abf);
  gemm_bf16_wmma<<<dim3(DMODEL / 64, PROWS / 128), 256, 0, stream>>>(
      abf, wpatch, ptmp, PROWS, DMODEL, KPATCH, patch_b, 0);
  assemble_tokens<<<cdiv((long long)ROWS * DMODEL, 256), 256, 0, stream>>>(
      ptmp, cls_tok, hs, res);

  const int gy = cdiv(ROWS, 128);
  for (int layer = 0; layer < NLAYERS; ++layer) {
    add_ln_bf16<<<ROWS, 128, 0, stream>>>(
        res, hs, ln_g + layer * DMODEL, ln_b + layer * DMODEL, abf);
    gemm_bf16_wmma<<<dim3(2 * EINNER / 64, gy), 256, 0, stream>>>(
        abf, win + (size_t)layer * 2 * EINNER * DMODEL, xz,
        ROWS, 2 * EINNER, DMODEL, nullptr, 0);
    conv_silu<<<cdiv((long long)ROWS * EINNER, 256), 256, 0, stream>>>(
        xz, conv_w + (size_t)layer * EINNER * DCONV, conv_b + layer * EINNER, xc, abf);
    gemm_bf16_wmma<<<dim3(cdiv(DBLW, 64), gy), 256, 0, stream>>>(
        abf, wx + (size_t)layer * DBLW * EINNER, dbl, ROWS, DBLW, EINNER, nullptr, 0);
    pad_bf16<<<cdiv((long long)ROWS * DTPAD, 256), 256, 0, stream>>>(
        dbl, abf, ROWS, DBLW, DTRANK);
    gemm_bf16_wmma<<<dim3(EINNER / 64, gy), 256, 0, stream>>>(
        abf, wdt + (size_t)layer * EINNER * DTPAD, dtv,
        ROWS, EINNER, DTPAD, dt_b + layer * EINNER, 1);
    selective_scan<<<BATCH * EINNER / 256, 256, 0, stream>>>(
        dtv, xc, dbl, xz, A_log + (size_t)layer * EINNER * NSTATE,
        D_skip + layer * EINNER, abf);
    gemm_bf16_wmma<<<dim3(DMODEL / 64, gy), 256, 0, stream>>>(
        abf, wout + (size_t)layer * DMODEL * EINNER, hs, ROWS, DMODEL, EINNER, nullptr, 0);
  }

  final_ln_cls<<<BATCH, 128, 0, stream>>>(res, hs, bb_g, bb_b, abf);
  gemm_bf16_wmma<<<dim3(cdiv(1000, 64), 1), 256, 0, stream>>>(
      abf, whead, out, BATCH, 1000, DMODEL, nullptr, 0);
}
